// CrossViewSwapAttention_6408091205811
// MI455X (gfx1250) — matrix operand order repaired and checked by reading, not compiled, not run on hardware
//
#include <hip/hip_runtime.h>
#include <hip/hip_bf16.h>
#include <math.h>

#define DEV __device__ __forceinline__

typedef __attribute__((ext_vector_type(16))) _Float16 v16h;
typedef __attribute__((ext_vector_type(8)))  float    v8f;

// ---- model constants ----
#define BB   2
#define NC   6
#define HGT  64
#define WID  64
#define FH   32
#define FW   88
#define DIM  128
#define HEADS 4
#define DHD  32
#define NQ   384      // NC * 8 * 8 query tokens per window
#define NK   264      // NC * 4 * 11 key tokens per window
#define NKP  288      // padded key count (multiple of 32)
#define EPSL 1e-5f

// =====================  WMMA helpers (CDNA5, wave32)  =====================

DEV v8f wmma16(v16h a, v16h b, v8f c) {
  // v_wmma_f32_16x16x32_f16
  return __builtin_amdgcn_wmma_f32_16x16x32_f16(false, a, false, b, (short)0, c,
                                                false, false);
}

// A-operand fragment: 16x32 f16. Rows = lane&15. K pattern per ISA 7.12.2:
// lanes 0-15: VGPR0..3 -> K 0..7, VGPR4..7 -> K 16..23 ; lanes 16-31: +8.
DEV v16h frag_a(const _Float16* base, int stride, int lane) {
  const _Float16* p = base + (lane & 15) * stride + ((lane & 16) ? 8 : 0);
  v16h f;
#pragma unroll
  for (int i = 0; i < 4; ++i) { f[2*i] = p[2*i]; f[2*i+1] = p[2*i+1]; }
#pragma unroll
  for (int i = 4; i < 8; ++i) { f[2*i] = p[2*i+8]; f[2*i+1] = p[2*i+9]; }
  return f;
}

// B-operand fragment loaded from B^T stored [N][K] row-major:
// lanes 0-15 hold K=0..7,16..23; lanes 16-31 hold K=8..15,24..31 (same as A).
DEV v16h frag_b(const _Float16* base, int stride, int lane) {
  const _Float16* p = base + (lane & 15) * stride + ((lane & 16) ? 8 : 0);
  v16h f;
#pragma unroll
  for (int i = 0; i < 16; ++i) f[i] = p[i < 8 ? i : i + 8];
  return f;
}

// ---- async global -> LDS copy (CDNA5 Tensor/async path, ASYNCcnt) ----
// Each active lane copies 16 bytes from its global address to its LDS offset.
DEV void async_copy_b128(unsigned lds_byte_off, const _Float16* g) {
  asm volatile("global_load_async_to_lds_b128 %0, %1, off"
               :: "v"(lds_byte_off), "v"(g)
               : "memory");
}
DEV void async_wait0() {
  asm volatile("s_wait_asynccnt 0x0" ::: "memory");
}
DEV unsigned lds_off(const void* p) {
  // generic pointers into LDS carry the group-relative offset in the low 32 bits
  return (unsigned)(unsigned long long)p;
}

// =====================  weight pre-conversion (f32 -> f16 B^T)  =============

// dst[o*K + i] = (f16)(trans ? src[i*N + o] : src[o*K + i])
__global__ void k_wconv(const float* __restrict__ src, _Float16* __restrict__ dst,
                        int N, int K, int trans) {
  int idx = blockIdx.x * 256 + threadIdx.x;
  if (idx >= N * K) return;
  int o = idx / K, i = idx % K;
  dst[idx] = (_Float16)(trans ? src[i * N + o] : src[idx]);
}

// =====================  preprocessing kernels  =====================

// c_embed[bn][o] = sum_j W_cam[o][j] * E_inv[bn][j][3]
__global__ void k_cembed(const float* __restrict__ Wcam,
                         const float* __restrict__ Einv,
                         float* __restrict__ cemb) {
  int bn = blockIdx.x, o = threadIdx.x;
  const float* E = Einv + bn * 16;
  float s = 0.f;
#pragma unroll
  for (int j = 0; j < 4; ++j) s += Wcam[o * 4 + j] * E[j * 4 + 3];
  cemb[bn * DIM + o] = s;
}

// img_embed -> key_buf (channel-last), later conv adds on top
__global__ void k_img_embed(const float* __restrict__ Iinv,
                            const float* __restrict__ Einv,
                            const float* __restrict__ Wimg,
                            const float* __restrict__ cemb,
                            float* __restrict__ key_buf) {
  __shared__ float red[DIM];
  int pix = blockIdx.x % (FH * FW);
  int bn  = blockIdx.x / (FH * FW);
  int h = pix / FW, w = pix % FW;
  int c = threadIdx.x;
  float gx = (float)w * (1.0f / (FW - 1)) * 480.0f;
  float gy = (float)h * (1.0f / (FH - 1)) * 224.0f;
  const float* I = Iinv + bn * 9;
  float c0 = I[0]*gx + I[1]*gy + I[2];
  float c1 = I[3]*gx + I[4]*gy + I[5];
  float c2 = I[6]*gx + I[7]*gy + I[8];
  const float* E = Einv + bn * 16;
  float d4[4];
#pragma unroll
  for (int i = 0; i < 4; ++i)
    d4[i] = E[i*4+0]*c0 + E[i*4+1]*c1 + E[i*4+2]*c2 + E[i*4+3];
  const float* wr = Wimg + c * 4;
  float v = wr[0]*d4[0] + wr[1]*d4[1] + wr[2]*d4[2] + wr[3]*d4[3]
            - cemb[bn * DIM + c];
  red[c] = v * v;
  __syncthreads();
  for (int s = 64; s > 0; s >>= 1) {
    if (c < s) red[c] += red[c + s];
    __syncthreads();
  }
  float dn = fmaxf(sqrtf(red[0]), 1e-12f);
  key_buf[((size_t)(bn * (FH*FW) + pix)) * DIM + c] = v / dn;
}

// bev_embed + x -> query_buf (b,n,H*W,128) channel-last
__global__ void k_bev_query(const float* __restrict__ grid,
                            const float* __restrict__ Wbev,
                            const float* __restrict__ bbev,
                            const float* __restrict__ cemb,
                            const float* __restrict__ x,
                            float* __restrict__ query_buf) {
  __shared__ float red[DIM];
  int pix = blockIdx.x % 4096;
  int bn  = blockIdx.x / 4096;
  int b = bn / NC;
  int c = threadIdx.x;
  float g0 = grid[pix], g1 = grid[4096 + pix];
  float v = Wbev[c*2+0]*g0 + Wbev[c*2+1]*g1 + bbev[c] - cemb[bn * DIM + c];
  red[c] = v * v;
  __syncthreads();
  for (int s = 64; s > 0; s >>= 1) {
    if (c < s) red[c] += red[c + s];
    __syncthreads();
  }
  float dn = fmaxf(sqrtf(red[0]), 1e-12f);
  query_buf[((size_t)(bn * 4096 + pix)) * DIM + c] =
      v / dn + x[(size_t)(b * DIM + c) * 4096 + pix];
}

// BN(eval) -> ReLU -> 1x1 conv (two of them) via WMMA.
// key_buf += W_fp conv ; val_buf = W_fl conv
__global__ void k_feat_conv(const float* __restrict__ feat,
                            const float* __restrict__ gp, const float* __restrict__ bp_,
                            const _Float16* __restrict__ cwfp,
                            const float* __restrict__ gl, const float* __restrict__ bl_,
                            const _Float16* __restrict__ cwfl,
                            float* __restrict__ key_buf, float* __restrict__ val_buf) {
  extern __shared__ char smem_raw[];
  _Float16* Ap  = (_Float16*)smem_raw;     // 64*128 relu act (key path)
  _Float16* Al  = Ap + 64 * DIM;           // 64*128 relu act (val path)
  _Float16* Wpt = Al + 64 * DIM;           // 128*128 f16 B^T
  _Float16* Wlt = Wpt + DIM * DIM;
  const int tid = threadIdx.x, wave = tid >> 5, lane = tid & 31;
  const int bn = blockIdx.x / 44;
  const int p0 = (blockIdx.x % 44) * 64;
  const float rs = rsqrtf(1.0f + EPSL);

  // async-DMA both weight matrices into LDS (no VGPR round trip)
  {
    unsigned wp_off = lds_off(Wpt), wl_off = lds_off(Wlt);
    for (int idx = tid; idx < 2048; idx += 128) {
      async_copy_b128(wp_off + idx * 16, cwfp + idx * 8);
      async_copy_b128(wl_off + idx * 16, cwfl + idx * 8);
    }
  }
  for (int idx = tid; idx < 64 * DIM; idx += 128) {
    int c = idx >> 6, t = idx & 63;
    float f = feat[((size_t)(bn * DIM + c)) * (FH*FW) + p0 + t];
    Ap[t * DIM + c] = (_Float16)fmaxf(f * rs * gp[c] + bp_[c], 0.f);
    Al[t * DIM + c] = (_Float16)fmaxf(f * rs * gl[c] + bl_[c], 0.f);
  }
  async_wait0();
  __syncthreads();

  const v8f vz = {0.f,0.f,0.f,0.f,0.f,0.f,0.f,0.f};
  v16h afp[4], afl[4];
#pragma unroll
  for (int ks = 0; ks < 4; ++ks) {
    afp[ks] = frag_a(Ap + wave*16*DIM + ks*32, DIM, lane);
    afl[ks] = frag_a(Al + wave*16*DIM + ks*32, DIM, lane);
  }
  for (int nt = 0; nt < 8; ++nt) {
    v8f ap = vz, al = vz;
#pragma unroll
    for (int ks = 0; ks < 4; ++ks) {
      v16h bwp = frag_b(Wpt + nt*16*DIM + ks*32, DIM, lane);
      v16h bwl = frag_b(Wlt + nt*16*DIM + ks*32, DIM, lane);
      ap = wmma16(afp[ks], bwp, ap);
      al = wmma16(afl[ks], bwl, al);
    }
    int n = nt * 16 + (lane & 15);
#pragma unroll
    for (int r = 0; r < 8; ++r) {
      int m = r + ((lane & 16) ? 8 : 0);
      size_t gi = ((size_t)(bn * (FH*FW) + p0 + wave*16 + m)) * DIM + n;
      key_buf[gi] += ap[r];
      val_buf[gi] = al[r];
    }
  }
}

// =====================  windowed cross attention  =====================

__global__ void k_attn(const float* __restrict__ qsrc, int use_q1,
                       const float* __restrict__ kbufg, const float* __restrict__ vbufg,
                       int kv_grid,
                       const float* __restrict__ lng, const float* __restrict__ lnb,
                       const _Float16* __restrict__ wqkv,   // f16 B^T: q,+16384 k,+32768 v
                       const float* __restrict__ bqkv,
                       const _Float16* __restrict__ wpt,    // f16 B^T out-proj
                       const float* __restrict__ bpo,
                       const float* __restrict__ skip, int skip_is_x,
                       float* __restrict__ obuf) {
  extern __shared__ char smem_raw[];
  _Float16* Qh = (_Float16*)smem_raw;      // 384*128 projected Q (later reused as O)
  _Float16* Kh = Qh + NQ * DIM;            // 288*128 projected K
  _Float16* Vt = Kh + NKP * DIM;           // 128*288 projected V (transposed)
  _Float16* Wt = Vt + DIM * NKP;           // 128*128 staged weight (B^T)
  _Float16* Xn = Wt + DIM * DIM;           // 4 * 16*128 per-wave LN tile
  _Float16* Pb = Xn + 4 * 16 * DIM;        // 4 * 16*32  per-wave P bounce

  const int tid = threadIdx.x, wave = tid >> 5, lane = tid & 31;
  const int b = blockIdx.x >> 6;
  const int l = blockIdx.x & 63;
  const int Xw = l >> 3, Yw = l & 7;
  const int row = lane >> 1, hf = lane & 1;
  const v8f vz = {0.f,0.f,0.f,0.f,0.f,0.f,0.f,0.f};
  const unsigned wt_off = lds_off(Wt);

  // async-stage Wq while we zero K/V pad region (keys 264..287)
  for (int idx = tid; idx < 2048; idx += 128)
    async_copy_b128(wt_off + idx * 16, wqkv + idx * 8);
  for (int idx = tid; idx < (NKP - NK) * DIM; idx += 128) {
    int r = idx / DIM, c = idx % DIM;
    Kh[(NK + r) * DIM + c] = (_Float16)0.f;
  }
  for (int idx = tid; idx < DIM * (NKP - NK); idx += 128) {
    int r = idx / (NKP - NK), c = idx % (NKP - NK);
    Vt[r * NKP + NK + c] = (_Float16)0.f;
  }
  async_wait0();
  __syncthreads();

  // ---------- Q projection ----------
  for (int t = wave; t < NQ / 16; t += 4) {
    int token = t * 16 + row;
    int cam = token >> 6, idx2 = token & 63;
    int i1 = idx2 >> 3, i2 = idx2 & 7;
    int h = Xw * 8 + i1, w = Yw * 8 + i2;
    const float* rp = use_q1
        ? qsrc + ((size_t)(b * 4096 + h * WID + w)) * DIM
        : qsrc + ((size_t)((b * NC + cam) * 4096 + h * WID + w)) * DIM;
    float s = 0.f, ss = 0.f;
    for (int j = 0; j < 64; ++j) { float v = rp[hf*64 + j]; s += v; ss += v*v; }
    s  += __shfl_xor(s, 1);
    ss += __shfl_xor(ss, 1);
    float mean = s * (1.f/128.f);
    float rstd = rsqrtf(ss * (1.f/128.f) - mean*mean + EPSL);
    _Float16* xr = Xn + wave*16*DIM + row*DIM;
    for (int j = 0; j < 64; ++j) {
      int c = hf*64 + j;
      xr[c] = (_Float16)(((rp[c] - mean) * rstd) * lng[c] + lnb[c]);
    }
    v16h afr[4];
#pragma unroll
    for (int ks = 0; ks < 4; ++ks)
      afr[ks] = frag_a(Xn + wave*16*DIM + ks*32, DIM, lane);
    for (int nt = 0; nt < 8; ++nt) {
      v8f acc = vz;
#pragma unroll
      for (int ks = 0; ks < 4; ++ks)
        acc = wmma16(afr[ks], frag_b(Wt + nt*16*DIM + ks*32, DIM, lane), acc);
      int n = nt * 16 + (lane & 15);
      float bi = bqkv[n];
#pragma unroll
      for (int r = 0; r < 8; ++r) {
        int m = r + ((lane & 16) ? 8 : 0);
        Qh[(t*16 + m) * DIM + n] = (_Float16)(acc[r] + bi);
      }
    }
  }
  __syncthreads();

  // ---------- K projection ----------
  for (int idx = tid; idx < 2048; idx += 128)
    async_copy_b128(wt_off + idx * 16, wqkv + DIM*DIM + idx * 8);
  async_wait0();
  __syncthreads();
  for (int t = wave; t < 17; t += 4) {
    int token = t * 16 + row;
    _Float16* xr = Xn + wave*16*DIM + row*DIM;
    if (token < NK) {
      int cam = token / 44, idx2 = token % 44;
      int j1 = idx2 / 11, j2 = idx2 % 11;
      int h = kv_grid ? (j1 * 8 + Xw) : (Xw * 4 + j1);
      int w = kv_grid ? (j2 * 8 + Yw) : (Yw * 11 + j2);
      const float* rp = kbufg + ((size_t)((b*NC + cam)*(FH*FW) + h*FW + w)) * DIM;
      float s = 0.f, ss = 0.f;
      for (int j = 0; j < 64; ++j) { float v = rp[hf*64 + j]; s += v; ss += v*v; }
      s  += __shfl_xor(s, 1);
      ss += __shfl_xor(ss, 1);
      float mean = s * (1.f/128.f);
      float rstd = rsqrtf(ss * (1.f/128.f) - mean*mean + EPSL);
      for (int j = 0; j < 64; ++j) {
        int c = hf*64 + j;
        xr[c] = (_Float16)(((rp[c] - mean) * rstd) * lng[DIM + c] + lnb[DIM + c]);
      }
    } else {
      for (int j = 0; j < 64; ++j) xr[hf*64 + j] = (_Float16)0.f;
    }
    v16h afr[4];
#pragma unroll
    for (int ks = 0; ks < 4; ++ks)
      afr[ks] = frag_a(Xn + wave*16*DIM + ks*32, DIM, lane);
    for (int nt = 0; nt < 8; ++nt) {
      v8f acc = vz;
#pragma unroll
      for (int ks = 0; ks < 4; ++ks)
        acc = wmma16(afr[ks], frag_b(Wt + nt*16*DIM + ks*32, DIM, lane), acc);
      int n = nt * 16 + (lane & 15);
      float bi = bqkv[DIM + n];
#pragma unroll
      for (int r = 0; r < 8; ++r) {
        int m = r + ((lane & 16) ? 8 : 0);
        Kh[(t*16 + m) * DIM + n] = (_Float16)(acc[r] + bi);
      }
    }
  }
  __syncthreads();

  // ---------- V projection (written transposed) ----------
  for (int idx = tid; idx < 2048; idx += 128)
    async_copy_b128(wt_off + idx * 16, wqkv + 2*DIM*DIM + idx * 8);
  async_wait0();
  __syncthreads();
  for (int t = wave; t < 17; t += 4) {
    int token = t * 16 + row;
    _Float16* xr = Xn + wave*16*DIM + row*DIM;
    if (token < NK) {
      int cam = token / 44, idx2 = token % 44;
      int j1 = idx2 / 11, j2 = idx2 % 11;
      int h = kv_grid ? (j1 * 8 + Xw) : (Xw * 4 + j1);
      int w = kv_grid ? (j2 * 8 + Yw) : (Yw * 11 + j2);
      const float* rp = vbufg + ((size_t)((b*NC + cam)*(FH*FW) + h*FW + w)) * DIM;
      float s = 0.f, ss = 0.f;
      for (int j = 0; j < 64; ++j) { float v = rp[hf*64 + j]; s += v; ss += v*v; }
      s  += __shfl_xor(s, 1);
      ss += __shfl_xor(ss, 1);
      float mean = s * (1.f/128.f);
      float rstd = rsqrtf(ss * (1.f/128.f) - mean*mean + EPSL);
      for (int j = 0; j < 64; ++j) {
        int c = hf*64 + j;
        xr[c] = (_Float16)(((rp[c] - mean) * rstd) * lng[2*DIM + c] + lnb[2*DIM + c]);
      }
    } else {
      for (int j = 0; j < 64; ++j) xr[hf*64 + j] = (_Float16)0.f;
    }
    v16h afr[4];
#pragma unroll
    for (int ks = 0; ks < 4; ++ks)
      afr[ks] = frag_a(Xn + wave*16*DIM + ks*32, DIM, lane);
    for (int nt = 0; nt < 8; ++nt) {
      v8f acc = vz;
#pragma unroll
      for (int ks = 0; ks < 4; ++ks)
        acc = wmma16(afr[ks], frag_b(Wt + nt*16*DIM + ks*32, DIM, lane), acc);
      int n = nt * 16 + (lane & 15);
      float bi = bqkv[2*DIM + n];
#pragma unroll
      for (int r = 0; r < 8; ++r) {
        int m = r + ((lane & 16) ? 8 : 0);
        Vt[n * NKP + (t*16 + m)] = (_Float16)(acc[r] + bi);
      }
    }
  }
  __syncthreads();

  // ---------- attention: units = (head, q-tile), two-pass online softmax ----
  {
    const float SCALE = 0.17677669529663687f;   // 1/sqrt(32)
    for (int unit = wave; unit < HEADS * (NQ/16); unit += 4) {
      int head = unit / (NQ/16);
      int qt   = unit % (NQ/16);
      v16h qa = frag_a(Qh + qt*16*DIM + head*DHD, DIM, lane);
      float mrow[8], lrow[8];
#pragma unroll
      for (int r = 0; r < 8; ++r) { mrow[r] = -1e30f; lrow[r] = 0.f; }
      // pass 1: row max / sum
      for (int kt = 0; kt < NKP/16; ++kt) {
        v16h kb = frag_b(Kh + kt*16*DIM + head*DHD, DIM, lane);
        v8f s = wmma16(qa, kb, vz);
        bool valid = (kt*16 + (lane & 15)) < NK;
#pragma unroll
        for (int r = 0; r < 8; ++r) {
          float sv = valid ? s[r] * SCALE : -1e30f;
          float tm = sv;
          tm = fmaxf(tm, __shfl_xor(tm, 1));
          tm = fmaxf(tm, __shfl_xor(tm, 2));
          tm = fmaxf(tm, __shfl_xor(tm, 4));
          tm = fmaxf(tm, __shfl_xor(tm, 8));
          float mn = fmaxf(mrow[r], tm);
          float e = __expf(sv - mn);
          e += __shfl_xor(e, 1);
          e += __shfl_xor(e, 2);
          e += __shfl_xor(e, 4);
          e += __shfl_xor(e, 8);
          lrow[r] = lrow[r] * __expf(mrow[r] - mn) + e;
          mrow[r] = mn;
        }
      }
      // pass 2: P = exp(S-m)/l, O += P @ V
      v8f acc0 = vz, acc1 = vz;
      _Float16* pb = Pb + wave * (16 * 32);
      for (int cc = 0; cc < NKP/32; ++cc) {
        for (int sub = 0; sub < 2; ++sub) {
          int kt = cc*2 + sub;
          v16h kb = frag_b(Kh + kt*16*DIM + head*DHD, DIM, lane);
          v8f s = wmma16(qa, kb, vz);
          bool valid = (kt*16 + (lane & 15)) < NK;
          int nn = lane & 15;
#pragma unroll
          for (int r = 0; r < 8; ++r) {
            float p = valid ? __expf(s[r]*SCALE - mrow[r]) / lrow[r] : 0.f;
            int m = r + ((lane & 16) ? 8 : 0);
            pb[m * 32 + sub*16 + nn] = (_Float16)p;
          }
        }
        v16h pa = frag_a(pb, 32, lane);
        acc0 = wmma16(pa, frag_b(Vt + (head*DHD)      * NKP + cc*32, NKP, lane), acc0);
        acc1 = wmma16(pa, frag_b(Vt + (head*DHD + 16) * NKP + cc*32, NKP, lane), acc1);
      }
      // write O in place (only clobbers the Q slice this unit consumed)
      int nn = lane & 15;
#pragma unroll
      for (int r = 0; r < 8; ++r) {
        int m = r + ((lane & 16) ? 8 : 0);
        int tok = qt*16 + m;
        Qh[tok*DIM + head*DHD + nn]      = (_Float16)acc0[r];
        Qh[tok*DIM + head*DHD + 16 + nn] = (_Float16)acc1[r];
      }
    }
  }
  __syncthreads();

  // ---------- output projection + camera mean + skip ----------
  for (int idx = tid; idx < 2048; idx += 128)
    async_copy_b128(wt_off + idx * 16, wpt + idx * 8);
  async_wait0();
  __syncthreads();
  {
    int rb = wave;   // row-block within window (16 of 64 positions)
    for (int nt = 0; nt < 8; ++nt) {
      v16h bfr[4];
#pragma unroll
      for (int ks = 0; ks < 4; ++ks)
        bfr[ks] = frag_b(Wt + nt*16*DIM + ks*32, DIM, lane);
      v8f acc = vz;
      for (int cam = 0; cam < NC; ++cam) {
        int base = cam * 64 + rb * 16;
#pragma unroll
        for (int ks = 0; ks < 4; ++ks)
          acc = wmma16(frag_a(Qh + base*DIM + ks*32, DIM, lane), bfr[ks], acc);
      }
      int n = nt * 16 + (lane & 15);
      float bi = bpo[n];
#pragma unroll
      for (int r = 0; r < 8; ++r) {
        int m = r + ((lane & 16) ? 8 : 0);
        int wr = rb * 16 + m;
        int i1 = wr >> 3, i2 = wr & 7;
        int h = Xw * 8 + i1, w = Yw * 8 + i2;
        float sk = skip_is_x
            ? skip[((size_t)(b*DIM + n) * HGT + h) * WID + w]
            : skip[((size_t)(b*4096 + h*WID + w)) * DIM + n];
        obuf[((size_t)(b*4096 + h*WID + w)) * DIM + n] =
            acc[r] * (1.f/6.f) + bi + sk;
      }
    }
  }
}

// =====================  MLP (LN -> 128->256 GELU -> 256->128 + residual) ====

__global__ void k_mlp(const float* __restrict__ in,
                      const float* __restrict__ g, const float* __restrict__ be,
                      const _Float16* __restrict__ w1t,   // 256x128 f16 B^T
                      const float* __restrict__ b1,
                      const _Float16* __restrict__ w2t,   // 128x256 f16 B^T
                      const float* __restrict__ b2,
                      float* __restrict__ out) {
  extern __shared__ char smem_raw[];
  _Float16* Xn  = (_Float16*)smem_raw;     // 64*128
  _Float16* W1t = Xn + 64 * DIM;           // 256*128
  _Float16* Hh  = W1t + 256 * DIM;         // 64*256
  _Float16* W2t = Hh + 64 * 256;           // 128*256
  const int tid = threadIdx.x, wave = tid >> 5, lane = tid & 31;
  const size_t tb = (size_t)blockIdx.x * 64;
  const v8f vz = {0.f,0.f,0.f,0.f,0.f,0.f,0.f,0.f};

  {
    unsigned w1_off = lds_off(W1t), w2_off = lds_off(W2t);
    for (int idx = tid; idx < 4096; idx += 128) {
      async_copy_b128(w1_off + idx * 16, w1t + idx * 8);
      async_copy_b128(w2_off + idx * 16, w2t + idx * 8);
    }
  }
  {
    int r = tid >> 1, hf = tid & 1;
    const float* rp = in + (tb + r) * DIM;
    float s = 0.f, ss = 0.f;
    for (int j = 0; j < 64; ++j) { float v = rp[hf*64 + j]; s += v; ss += v*v; }
    s  += __shfl_xor(s, 1);
    ss += __shfl_xor(ss, 1);
    float mean = s * (1.f/128.f);
    float rstd = rsqrtf(ss * (1.f/128.f) - mean*mean + EPSL);
    for (int j = 0; j < 64; ++j) {
      int c = hf*64 + j;
      Xn[r * DIM + c] = (_Float16)(((rp[c] - mean) * rstd) * g[c] + be[c]);
    }
  }
  async_wait0();
  __syncthreads();

  v16h afr[4];
#pragma unroll
  for (int ks = 0; ks < 4; ++ks)
    afr[ks] = frag_a(Xn + wave*16*DIM + ks*32, DIM, lane);
  for (int nt = 0; nt < 16; ++nt) {
    v8f acc = vz;
#pragma unroll
    for (int ks = 0; ks < 4; ++ks)
      acc = wmma16(afr[ks], frag_b(W1t + nt*16*DIM + ks*32, DIM, lane), acc);
    int n = nt * 16 + (lane & 15);
    float bi = b1[n];
#pragma unroll
    for (int r = 0; r < 8; ++r) {
      int m = r + ((lane & 16) ? 8 : 0);
      float xv = acc[r] + bi;
      Hh[(wave*16 + m) * 256 + n] =
          (_Float16)(0.5f * xv * (1.f + erff(xv * 0.70710678118f)));
    }
  }
  v16h hfr[8];
#pragma unroll
  for (int ks = 0; ks < 8; ++ks)
    hfr[ks] = frag_a(Hh + wave*16*256 + ks*32, 256, lane);
  for (int nt = 0; nt < 8; ++nt) {
    v8f acc = vz;
#pragma unroll
    for (int ks = 0; ks < 8; ++ks)
      acc = wmma16(hfr[ks], frag_b(W2t + nt*16*256 + ks*32, 256, lane), acc);
    int n = nt * 16 + (lane & 15);
    float bi = b2[n];
#pragma unroll
    for (int r = 0; r < 8; ++r) {
      int m = r + ((lane & 16) ? 8 : 0);
      size_t tok = tb + wave*16 + m;
      out[tok * DIM + n] = in[tok * DIM + n] + acc[r] + bi;
    }
  }
}

// =====================  final LN + transpose to (b, c, h, w)  ===============

__global__ void k_postln(const float* __restrict__ q2,
                         const float* __restrict__ g, const float* __restrict__ be,
                         float* __restrict__ out) {
  __shared__ float r1[DIM], r2[DIM];
  int tok = blockIdx.x, c = threadIdx.x;
  float v = q2[(size_t)tok * DIM + c];
  r1[c] = v; r2[c] = v * v;
  __syncthreads();
  for (int s = 64; s > 0; s >>= 1) {
    if (c < s) { r1[c] += r1[c + s]; r2[c] += r2[c + s]; }
    __syncthreads();
  }
  float mean = r1[0] * (1.f/128.f);
  float rstd = rsqrtf(r2[0] * (1.f/128.f) - mean*mean + EPSL);
  int b = tok >> 12, pix = tok & 4095;
  out[((size_t)(b * DIM + c)) * 4096 + pix] = (v - mean) * rstd * g[c] + be[c];
}

// =====================  launch  =====================

extern "C" void kernel_launch(void* const* d_in, const int* in_sizes, int n_in,
                              void* d_out, int out_size, void* d_ws, size_t ws_size,
                              hipStream_t stream) {
  (void)in_sizes; (void)n_in; (void)out_size; (void)ws_size;
  const float* x       = (const float*)d_in[1];
  const float* grid    = (const float*)d_in[2];
  const float* feature = (const float*)d_in[3];
  const float* Iinv    = (const float*)d_in[4];
  const float* Einv    = (const float*)d_in[5];
  const float* bn_fl_g = (const float*)d_in[6];
  const float* bn_fl_b = (const float*)d_in[7];
  const float* W_fl    = (const float*)d_in[8];
  const float* bn_fp_g = (const float*)d_in[9];
  const float* bn_fp_b = (const float*)d_in[10];
  const float* W_fp    = (const float*)d_in[11];
  const float* W_bev   = (const float*)d_in[12];
  const float* b_bev   = (const float*)d_in[13];
  const float* W_img   = (const float*)d_in[14];
  const float* W_cam   = (const float*)d_in[15];
  const float* ln_g    = (const float*)d_in[16];
  const float* ln_b    = (const float*)d_in[17];
  const float* Wqkv    = (const float*)d_in[18];
  const float* bqkv    = (const float*)d_in[19];
  const float* Wp      = (const float*)d_in[20];
  const float* bp      = (const float*)d_in[21];
  const float* pn_g    = (const float*)d_in[22];
  const float* pn_b    = (const float*)d_in[23];
  const float* Wma     = (const float*)d_in[24];
  const float* bma     = (const float*)d_in[25];
  const float* Wmb     = (const float*)d_in[26];
  const float* bmb     = (const float*)d_in[27];
  const float* post_g  = (const float*)d_in[28];
  const float* post_b  = (const float*)d_in[29];
  float* outp = (float*)d_out;

  // workspace layout (floats)
  float* wsf     = (float*)d_ws;
  float* c_emb   = wsf;                               // 1536
  float* key_buf = wsf + 4096;                        // 12*2816*128
  float* val_buf = key_buf + 12 * (FH*FW) * DIM;
  float* query   = val_buf + 12 * (FH*FW) * DIM;      // 12*4096*128
  float* o_buf   = query + 12 * 4096 * DIM;           // 2*4096*128
  float* q1_buf  = o_buf + BB * 4096 * DIM;
  float* q2_buf  = q1_buf + BB * 4096 * DIM;
  // f16 pre-converted weights (B^T layout), 16-byte aligned tail region
  _Float16* wf16 = (_Float16*)(q2_buf + BB * 4096 * DIM);
  _Float16* cw_fp = wf16;                    // 128x128
  _Float16* cw_fl = cw_fp + DIM * DIM;       // 128x128
  _Float16* aw0   = cw_fl + DIM * DIM;       // stage0: q,k,v,p  (4 x 128x128)
  _Float16* aw1   = aw0 + 4 * DIM * DIM;     // stage1: q,k,v,p
  _Float16* mw1_0 = aw1 + 4 * DIM * DIM;     // stage0 MLP W1 (256x128)
  _Float16* mw2_0 = mw1_0 + 256 * DIM;       // stage0 MLP W2 (128x256)
  _Float16* mw1_1 = mw2_0 + DIM * 256;       // stage1 MLP W1
  _Float16* mw2_1 = mw1_1 + 256 * DIM;       // stage1 MLP W2

  const int ATTN_LDS = (NQ*DIM + NKP*DIM + DIM*NKP + DIM*DIM + 4*16*DIM + 4*16*32) * 2;
  const int CONV_LDS = (64*DIM*2 + DIM*DIM*2) * 2;
  const int MLP_LDS  = (64*DIM + 256*DIM + 64*256 + DIM*256) * 2;

  // ---- one-time weight conversion to f16 B^T ----
  k_wconv<<<64, 256, 0, stream>>>(W_fp, cw_fp, DIM, DIM, 0);
  k_wconv<<<64, 256, 0, stream>>>(W_fl, cw_fl, DIM, DIM, 0);
  for (int s = 0; s < 2; ++s) {
    _Float16* aw = s ? aw1 : aw0;
    for (int which = 0; which < 3; ++which)
      k_wconv<<<64, 256, 0, stream>>>(Wqkv + (s*3 + which) * DIM * DIM,
                                      aw + which * DIM * DIM, DIM, DIM, 1);
    k_wconv<<<64, 256, 0, stream>>>(Wp + s * DIM * DIM, aw + 3 * DIM * DIM,
                                    DIM, DIM, 1);
    k_wconv<<<128, 256, 0, stream>>>(Wma + s * DIM * 256, s ? mw1_1 : mw1_0,
                                     256, DIM, 1);
    k_wconv<<<128, 256, 0, stream>>>(Wmb + s * 256 * DIM, s ? mw2_1 : mw2_0,
                                     DIM, 256, 1);
  }

  // ---- preprocessing ----
  k_cembed<<<BB*NC, DIM, 0, stream>>>(W_cam, Einv, c_emb);
  k_img_embed<<<BB*NC*FH*FW, DIM, 0, stream>>>(Iinv, Einv, W_img, c_emb, key_buf);
  k_bev_query<<<BB*NC*4096, DIM, 0, stream>>>(grid, W_bev, b_bev, c_emb, x, query);
  k_feat_conv<<<BB*NC*44, 128, CONV_LDS, stream>>>(feature,
      bn_fp_g, bn_fp_b, cw_fp, bn_fl_g, bn_fl_b, cw_fl, key_buf, val_buf);

  // ---- stage 1: local windows, skip = x (channel-major) ----
  k_attn<<<BB*64, 128, ATTN_LDS, stream>>>(query, 0, key_buf, val_buf, 0,
      ln_g, ln_b, aw0, bqkv, aw0 + 3*DIM*DIM, bp, x, 1, o_buf);
  k_mlp<<<BB*64, 128, MLP_LDS, stream>>>(o_buf, pn_g, pn_b,
      mw1_0, bma, mw2_0, bmb, q1_buf);

  // ---- stage 2: grid windows, q broadcast from q1, skip = q1 ----
  k_attn<<<BB*64, 128, ATTN_LDS, stream>>>(q1_buf, 1, key_buf, val_buf, 1,
      ln_g + 3*DIM, ln_b + 3*DIM, aw1, bqkv + 3*DIM,
      aw1 + 3*DIM*DIM, bp + DIM, q1_buf, 0, o_buf);
  k_mlp<<<BB*64, 128, MLP_LDS, stream>>>(o_buf, pn_g + DIM, pn_b + DIM,
      mw1_1, bma + 256, mw2_1, bmb + DIM, q2_buf);

  k_postln<<<BB*4096, DIM, 0, stream>>>(q2_buf, post_g, post_b, outp);
}
